// Model_6442450944409
// MI455X (gfx1250) — matrix operand order repaired and checked by reading, not compiled, not run on hardware
//
#include <hip/hip_runtime.h>

// ---------------------------------------------------------------------------
// Model dims
// ---------------------------------------------------------------------------
#define SEQL 10
#define NCN  7
#define NNN  10
#define GHD  32
#define HEADS4 4
#define EDM  128
#define NEGV (-1000000000.0f)

typedef __attribute__((ext_vector_type(16))) _Float16 v16h;
typedef __attribute__((ext_vector_type(8)))  _Float16 v8h;
typedef __attribute__((ext_vector_type(8)))  float    v8f;

union H16 { v16h v; v8h h[2]; };

// ---------------------------------------------------------------------------
// Fast transcendentals: CDNA5 has V_TANH_F32; use it if the builtin exists.
// exp via raw v_exp (exp2) -- inputs are max-subtracted so no range fixup.
// ---------------------------------------------------------------------------
#if __has_builtin(__builtin_amdgcn_tanhf)
__device__ __forceinline__ float fast_tanh(float x) {
    return __builtin_amdgcn_tanhf(x);
}
#else
__device__ __forceinline__ float fast_tanh(float x) { return tanhf(x); }
#endif

__device__ __forceinline__ float fast_sig(float x) {
    return 0.5f * fast_tanh(0.5f * x) + 0.5f;  // sigma(x) = (1+tanh(x/2))/2
}

#define LOG2E 1.4426950408889634f
__device__ __forceinline__ float fast_exp(float x) {
    return __builtin_amdgcn_exp2f(x * LOG2E);
}

// ---------------------------------------------------------------------------
// WMMA helpers (CDNA5 16x16x32 f16 -> f32)
// A-frag (16x32 f16) from LDS row-major (stride in halves):
//   lane<16 : row=lane,  halves = K[0..7]  and K[16..23]
//   lane>=16: row=l-16,  halves = K[8..15] and K[24..31]
// B-frag (32x16 f16) from row-major weights W[outCol][K]:
//   lane col = l&15, halves = K[8h..8h+7] and K[16+8h..16+8h+7], h = l>>4
// ---------------------------------------------------------------------------
__device__ __forceinline__ v16h load_afrag_lds(const _Float16* row0,
                                               int stride_halves, int lane) {
    int r  = lane & 15;
    int hi = (lane >> 4) & 1;
    const _Float16* p = row0 + r * stride_halves + hi * 8;
    H16 u;
    u.h[0] = *(const v8h*)(p);
    u.h[1] = *(const v8h*)(p + 16);
    return u.v;
}

__device__ __forceinline__ v16h load_bfrag_g(const _Float16* w, int n0,
                                             int rowlen, int k0, int lane) {
    int n  = lane & 15;
    int hi = (lane >> 4) & 1;
    const _Float16* p = w + (size_t)(n0 + n) * rowlen + k0 + hi * 8;
    H16 u;
    u.h[0] = *(const v8h*)(p);
    u.h[1] = *(const v8h*)(p + 16);
    return u.v;
}

__device__ __forceinline__ v8f wmma_f16(v16h a, v16h b, v8f c) {
    return __builtin_amdgcn_wmma_f32_16x16x32_f16(false, a, false, b,
                                                  (short)0, c, false, false);
}

// ---------------------------------------------------------------------------
// Kernel 1: weight prep (f32 -> f16, pad Wih1 K 21->32, fuse biases)
// ---------------------------------------------------------------------------
struct PrepArgs {
    const float* l1Wih[2]; const float* l1Whh[2];
    const float* l1bih[2]; const float* l1bhh[2];
    const float* l2Wih[2]; const float* l2Whh[2];
    const float* l2bih[2]; const float* l2bhh[2];
    const float* a1W; const float* a2W; const float* fcW;
    _Float16* W1ih; _Float16* W1hh; float* B1;
    _Float16* W2ih; _Float16* W2hh; float* B2;
    _Float16* A1W;  _Float16* A2W;  _Float16* FCW;
};

__global__ void prep_kernel(PrepArgs A) {
    int tid    = blockIdx.x * blockDim.x + threadIdx.x;
    int stride = gridDim.x * blockDim.x;
    for (int d = 0; d < 2; ++d) {
        for (int i = tid; i < 256 * 32; i += stride) {
            int r = i >> 5, k = i & 31;
            A.W1ih[d * 256 * 32 + i] =
                (k < 21) ? (_Float16)A.l1Wih[d][r * 21 + k] : (_Float16)0.f;
        }
        for (int i = tid; i < 256 * 64; i += stride)
            A.W1hh[d * 256 * 64 + i] = (_Float16)A.l1Whh[d][i];
        for (int i = tid; i < 256; i += stride)
            A.B1[d * 256 + i] = A.l1bih[d][i] + A.l1bhh[d][i];
        for (int i = tid; i < 256 * 128; i += stride)
            A.W2ih[d * 256 * 128 + i] = (_Float16)A.l2Wih[d][i];
        for (int i = tid; i < 256 * 64; i += stride)
            A.W2hh[d * 256 * 64 + i] = (_Float16)A.l2Whh[d][i];
        for (int i = tid; i < 256; i += stride)
            A.B2[d * 256 + i] = A.l2bih[d][i] + A.l2bhh[d][i];
    }
    for (int i = tid; i < 128 * 128; i += stride) {
        A.A1W[i] = (_Float16)A.a1W[i];
        A.A2W[i] = (_Float16)A.a2W[i];
        A.FCW[i] = (_Float16)A.fcW[i];
    }
}

// ---------------------------------------------------------------------------
// Kernel 2: graph front-end (GAT1 -> GAT2, gconv1 -> gconv2).
// One wave per sample, 8 samples per workgroup. Writes X (B,10,21) f32.
// ---------------------------------------------------------------------------
__global__ __launch_bounds__(256) void graph_kernel(
    const float* __restrict__ M,
    const float* __restrict__ ga1W, const float* __restrict__ ga1s,
    const float* __restrict__ ga1d, const float* __restrict__ ga1b,
    const float* __restrict__ ga2W, const float* __restrict__ ga2s,
    const float* __restrict__ ga2d, const float* __restrict__ ga2b,
    const float* __restrict__ gp1rel, const float* __restrict__ gp1root,
    const float* __restrict__ gp1b,
    const float* __restrict__ gp2rel, const float* __restrict__ gp2root,
    const float* __restrict__ gp2b,
    float* __restrict__ X) {
    __shared__ float x0s[8][10][10];
    __shared__ float xp2s[8][10][10];

    const int lane = threadIdx.x & 31;
    const int wave = threadIdx.x >> 5;
    const size_t b = (size_t)blockIdx.x * 8 + wave;

    const int adjSL[10] = {0x081, 0x082, 0x084, 0x088, 0x110,
                           0x120, 0x140, 0x38F, 0x3F0, 0x380};
    const int adj[10]   = {0x080, 0x080, 0x080, 0x080, 0x100,
                           0x100, 0x100, 0x30F, 0x2F0, 0x180};

    // x0 = concat(M, zeros) -> LDS
    for (int i = lane; i < 100; i += 32) {
        int n = i / 10, f = i % 10;
        x0s[wave][n][f] = (n < 7) ? M[b * 70 + n * 10 + f] : 0.f;
    }
    __syncthreads();

    // ---- GAT1: xp[n][h] with h = register slot, g = lane -------------------
    float xp[10][4];
#pragma unroll
    for (int n = 0; n < 10; ++n)
#pragma unroll
        for (int h = 0; h < 4; ++h) {
            float acc = 0.f;
#pragma unroll
            for (int f = 0; f < 10; ++f)
                acc += x0s[wave][n][f] * ga1W[(h * 32 + lane) * 10 + f];
            xp[n][h] = acc;
        }

    float sA[10][4], dA[10][4];
#pragma unroll
    for (int n = 0; n < 10; ++n)
#pragma unroll
        for (int h = 0; h < 4; ++h) {
            float ps = xp[n][h] * ga1s[h * 32 + lane];
            float pd = xp[n][h] * ga1d[h * 32 + lane];
            for (int off = 16; off; off >>= 1) {
                ps += __shfl_xor(ps, off);
                pd += __shfl_xor(pd, off);
            }
            sA[n][h] = ps;
            dA[n][h] = pd;
        }

    float h1[10][4];
#pragma unroll
    for (int i = 0; i < 10; ++i)
#pragma unroll
        for (int h = 0; h < 4; ++h) {
            float lg[10], mx = -3.4e38f;
#pragma unroll
            for (int j = 0; j < 10; ++j) {
                float v = dA[i][h] + sA[j][h];
                v       = (v < 0.f) ? 0.2f * v : v;
                lg[j]   = ((adjSL[i] >> j) & 1) ? v : NEGV;
                mx      = fmaxf(mx, lg[j]);
            }
            float sum = 0.f;
#pragma unroll
            for (int j = 0; j < 10; ++j) {
                lg[j] = fast_exp(lg[j] - mx);
                sum += lg[j];
            }
            float inv = 1.f / sum, o = 0.f;
#pragma unroll
            for (int j = 0; j < 10; ++j) o += lg[j] * xp[j][h];
            o        = o * inv + ga1b[h * 32 + lane];
            h1[i][h] = fmaxf(o, 0.f);
        }

    // ---- GAT2: project 128 -> 10 (wave reductions), attention head 0 -------
#pragma unroll
    for (int n = 0; n < 10; ++n)
#pragma unroll
        for (int f = 0; f < 10; ++f) {
            float p = 0.f;
#pragma unroll
            for (int h = 0; h < 4; ++h)
                p += h1[n][h] * ga2W[f * 128 + h * 32 + lane];
            for (int off = 16; off; off >>= 1) p += __shfl_xor(p, off);
            if (lane == 0) xp2s[wave][n][f] = p;
        }
    __syncthreads();

    float s2[10], d2[10];
#pragma unroll
    for (int n = 0; n < 10; ++n) {
        float as = 0.f, ad = 0.f;
#pragma unroll
        for (int f = 0; f < 10; ++f) {
            as += xp2s[wave][n][f] * ga2s[f];
            ad += xp2s[wave][n][f] * ga2d[f];
        }
        s2[n] = as;
        d2[n] = ad;
    }

#pragma unroll
    for (int i = 0; i < 7; ++i) {
        float lg[10], mx = -3.4e38f;
#pragma unroll
        for (int j = 0; j < 10; ++j) {
            float v = d2[i] + s2[j];
            v       = (v < 0.f) ? 0.2f * v : v;
            lg[j]   = ((adjSL[i] >> j) & 1) ? v : NEGV;
            mx      = fmaxf(mx, lg[j]);
        }
        float sum = 0.f;
#pragma unroll
        for (int j = 0; j < 10; ++j) {
            lg[j] = fast_exp(lg[j] - mx);
            sum += lg[j];
        }
        float inv = 1.f / sum;
        if (lane < 10) {
            float mp = 0.f;
#pragma unroll
            for (int j = 0; j < 10; ++j) mp += lg[j] * xp2s[wave][j][lane];
            mp = mp * inv + ga2b[lane];
            X[b * 210 + lane * 21 + 7 + i] = mp;  // x[b,t,7+i] = Mp[b,i,t]
        }
    }

    // ---- gconv path: k = lane (GH=32) --------------------------------------
    float g[10];
#pragma unroll
    for (int n = 0; n < 10; ++n) {
        float accR = 0.f, accA = 0.f;
#pragma unroll
        for (int f = 0; f < 10; ++f) {
            float xa = 0.f;
#pragma unroll
            for (int j = 0; j < 10; ++j)
                if ((adj[n] >> j) & 1) xa += x0s[wave][j][f];
            accR += x0s[wave][n][f] * gp1root[lane * 10 + f];
            accA += xa * gp1rel[lane * 10 + f];
        }
        g[n] = fmaxf(accR + accA + gp1b[lane], 0.f);
    }
    float ag[10];
#pragma unroll
    for (int n = 0; n < 10; ++n) {
        float t = 0.f;
#pragma unroll
        for (int j = 0; j < 10; ++j)
            if ((adj[n] >> j) & 1) t += g[j];
        ag[n] = t;
    }
#pragma unroll
    for (int i = 0; i < 7; ++i)
#pragma unroll
        for (int s = 0; s < 10; ++s) {
            float p = g[i] * gp2root[s * 32 + lane] + ag[i] * gp2rel[s * 32 + lane];
            for (int off = 16; off; off >>= 1) p += __shfl_xor(p, off);
            if (lane == 0)
                X[b * 210 + s * 21 + 14 + i] = p + gp2b[s];  // x[b,t,14+i]
        }

    // ---- raw M part: x[b,t,n] = M[b,n,t] ----------------------------------
    for (int idx = lane; idx < 70; idx += 32) {
        int n = idx / 10, t = idx % 10;
        X[b * 210 + t * 21 + n] = M[b * 70 + idx];
    }
}

// ---------------------------------------------------------------------------
// Kernel 3: fused BiLSTM1 -> attn1 -> BiLSTM2 -> attn2 -> mean -> FC.
// 16 samples per workgroup, 8 waves. Wave = (dir, hidden-slice-of-16).
// ---------------------------------------------------------------------------
#define OSTR 136  // padded row stride (halves) for 128-wide f16 LDS matrices
#define XSTR 40   // padded row stride (halves) for 32-wide f16 LDS matrix

__device__ void run_bilstm(const _Float16* Xin, int xstride, int nkx,
                           const _Float16* Wih, int ihrow,
                           const _Float16* Whh, const float* Bc, _Float16* O,
                           int lane, int wave) {
    const int d  = wave >> 2;  // 0=fwd, 1=bwd
    const int hs = wave & 3;   // 16-wide hidden slice
    const _Float16* wih = Wih + (size_t)d * 256 * ihrow;
    const _Float16* whh = Whh + (size_t)d * 256 * 64;
    const float*    bc  = Bc + d * 256;
    const int n  = hs * 16 + (lane & 15);
    const int hi = (lane >> 4) & 1;

    float cst[8];
#pragma unroll
    for (int v = 0; v < 8; ++v) cst[v] = 0.f;

    for (int k = 0; k < 10; ++k) {
        const int t = d ? (9 - k) : k;
        v8f acc[4];
#pragma unroll
        for (int q = 0; q < 4; ++q) {
            float bq = bc[q * 64 + n];
#pragma unroll
            for (int v = 0; v < 8; ++v) acc[q][v] = bq;
        }
        // input part
        for (int kf = 0; kf < nkx; ++kf) {
            v16h ax = load_afrag_lds(Xin + (size_t)t * 16 * xstride + kf * 32,
                                     xstride, lane);
#pragma unroll
            for (int q = 0; q < 4; ++q) {
                v16h bw = load_bfrag_g(wih, q * 64 + hs * 16, ihrow, kf * 32, lane);
                acc[q]  = wmma_f16(ax, bw, acc[q]);
            }
        }
        // recurrent part
        if (k > 0) {
            const int tp = d ? (t + 1) : (t - 1);
            for (int kf = 0; kf < 2; ++kf) {
                v16h ah = load_afrag_lds(
                    O + (size_t)tp * 16 * OSTR + d * 64 + kf * 32, OSTR, lane);
#pragma unroll
                for (int q = 0; q < 4; ++q) {
                    v16h bw = load_bfrag_g(whh, q * 64 + hs * 16, 64, kf * 32, lane);
                    acc[q]  = wmma_f16(ah, bw, acc[q]);
                }
            }
        }
        // cell update (c lives in VGPRs across steps); hardware tanh path
#pragma unroll
        for (int v = 0; v < 8; ++v) {
            float ig = fast_sig(acc[0][v]);
            float fg = fast_sig(acc[1][v]);
            float gg = fast_tanh(acc[2][v]);
            float og = fast_sig(acc[3][v]);
            cst[v]   = fg * cst[v] + ig * gg;
            float h  = og * fast_tanh(cst[v]);
            int m    = v + 8 * hi;
            O[((size_t)t * 16 + m) * OSTR + d * 64 + n] = (_Float16)h;
        }
        __syncthreads();
    }
}

__device__ void attn_q(const _Float16* O, _Float16* Q, const _Float16* W,
                       const float* bias, int lane, int wave) {
    const int n0 = wave * 16;
    const int hi = (lane >> 4) & 1;
    const int n  = n0 + (lane & 15);
    for (int mt = 0; mt < 10; ++mt) {
        v8f acc;
        float bq = bias[n];
#pragma unroll
        for (int v = 0; v < 8; ++v) acc[v] = bq;
        for (int kf = 0; kf < 4; ++kf) {
            v16h a  = load_afrag_lds(O + (size_t)mt * 16 * OSTR + kf * 32, OSTR, lane);
            v16h bw = load_bfrag_g(W, n0, 128, kf * 32, lane);
            acc     = wmma_f16(a, bw, acc);
        }
#pragma unroll
        for (int v = 0; v < 8; ++v) {
            int m = v + 8 * hi;
            Q[((size_t)mt * 16 + m) * OSTR + n] = (_Float16)fast_tanh(acc[v]);
        }
    }
    __syncthreads();
}

__device__ void attn_scores(const _Float16* Q, const _Float16* O, float* SC,
                            int tid) {
    const int s = tid >> 4, r = tid & 15;
    for (int p = r; p < 100; p += 16) {
        int i = p / 10, j = p % 10;
        const _Float16* q = Q + ((size_t)i * 16 + s) * OSTR;
        const _Float16* o = O + ((size_t)j * 16 + s) * OSTR;
        float acc = 0.f;
        for (int c = 0; c < 128; ++c) acc += (float)q[c] * (float)o[c];
        SC[(s * 10 + i) * 10 + j] = acc;
    }
    __syncthreads();
    if (r < 10) {
        float* row = SC + (s * 10 + r) * 10;
        float mx = row[0];
#pragma unroll
        for (int j = 1; j < 10; ++j) mx = fmaxf(mx, row[j]);
        float sum = 0.f, e[10];
#pragma unroll
        for (int j = 0; j < 10; ++j) { e[j] = fast_exp(row[j] - mx); sum += e[j]; }
        float inv = 1.f / sum;
#pragma unroll
        for (int j = 0; j < 10; ++j) row[j] = e[j] * inv;
    }
    __syncthreads();
}

__device__ void attn_apply(const float* SC, const _Float16* O, _Float16* D,
                           int tid) {
    const int s = tid >> 4, r = tid & 15;
    const int c0 = r * 8;
    for (int i = 0; i < 10; ++i) {
        float acc[8];
#pragma unroll
        for (int c = 0; c < 8; ++c) acc[c] = 0.f;
        for (int j = 0; j < 10; ++j) {
            float a = SC[(s * 10 + i) * 10 + j];
            const _Float16* o = O + ((size_t)j * 16 + s) * OSTR + c0;
#pragma unroll
            for (int c = 0; c < 8; ++c) acc[c] += a * (float)o[c];
        }
        _Float16* d = D + ((size_t)i * 16 + s) * OSTR + c0;
#pragma unroll
        for (int c = 0; c < 8; ++c) d[c] = (_Float16)acc[c];
    }
    __syncthreads();
}

__device__ void attn_mean_apply(const float* SC, const _Float16* O,
                                _Float16* MB, float* AB, int tid) {
    const int s = tid >> 4, r = tid & 15;
    if (r < 10) {  // mean over i commutes into the weights
        float a = 0.f;
        for (int i = 0; i < 10; ++i) a += SC[(s * 10 + i) * 10 + r];
        AB[s * 10 + r] = a * 0.1f;
    }
    __syncthreads();
    const int c0 = r * 8;
    float acc[8];
#pragma unroll
    for (int c = 0; c < 8; ++c) acc[c] = 0.f;
    for (int j = 0; j < 10; ++j) {
        float a = AB[s * 10 + j];
        const _Float16* o = O + ((size_t)j * 16 + s) * OSTR + c0;
#pragma unroll
        for (int c = 0; c < 8; ++c) acc[c] += a * (float)o[c];
    }
#pragma unroll
    for (int c = 0; c < 8; ++c) MB[s * OSTR + c0 + c] = (_Float16)acc[c];
    __syncthreads();
}

__global__ __launch_bounds__(256) void fused_seq_kernel(
    const float* __restrict__ X,
    const _Float16* __restrict__ W1ih, const _Float16* __restrict__ W1hh,
    const float* __restrict__ B1,
    const _Float16* __restrict__ W2ih, const _Float16* __restrict__ W2hh,
    const float* __restrict__ B2,
    const _Float16* __restrict__ A1W, const float* __restrict__ A1b,
    const _Float16* __restrict__ A2W, const float* __restrict__ A2b,
    const _Float16* __restrict__ FCW, const float* __restrict__ FCb,
    float* __restrict__ out) {
    __shared__ __attribute__((aligned(16))) _Float16 Xl[10 * 16 * XSTR];
    __shared__ __attribute__((aligned(16))) _Float16 O1[10 * 16 * OSTR];
    __shared__ __attribute__((aligned(16))) _Float16 QX[10 * 16 * OSTR];
    __shared__ float SC[16 * 10 * 10];
    __shared__ float AB[16 * 10];
    __shared__ __attribute__((aligned(16))) _Float16 MB[16 * OSTR];

    const int tid  = threadIdx.x;
    const int lane = tid & 31;
    const int wave = tid >> 5;
    const size_t b0 = (size_t)blockIdx.x * 16;

    // stage LSTM1 input (f32 -> f16, zero-pad K 21 -> 32)
    for (int i = tid; i < 10 * 16 * XSTR; i += 256) Xl[i] = (_Float16)0.f;
    __syncthreads();
    for (int i = tid; i < 16 * 210; i += 256) {
        int s = i / 210, rem = i % 210, t = rem / 21, f = rem % 21;
        Xl[(t * 16 + s) * XSTR + f] = (_Float16)X[(b0 + s) * 210 + rem];
    }
    __syncthreads();

    run_bilstm(Xl, XSTR, 1, W1ih, 32, W1hh, B1, O1, lane, wave);   // O1 = BiLSTM1
    attn_q(O1, QX, A1W, A1b, lane, wave);                          // Q1
    attn_scores(QX, O1, SC, tid);                                  // softmax
    attn_apply(SC, O1, QX, tid);                                   // X2 -> QX

    run_bilstm(QX, OSTR, 4, W2ih, 128, W2hh, B2, O1, lane, wave);  // O2 -> O1
    attn_q(O1, QX, A2W, A2b, lane, wave);                          // Q2 -> QX
    attn_scores(QX, O1, SC, tid);
    attn_mean_apply(SC, O1, MB, AB, tid);                          // mean -> MB

    // final FC via WMMA: out(16x128) = MB(16x128) @ FCW^T + FCb
    {
        const int n0 = wave * 16;
        const int hi = (lane >> 4) & 1;
        const int n  = n0 + (lane & 15);
        v8f acc;
        float bq = FCb[n];
#pragma unroll
        for (int v = 0; v < 8; ++v) acc[v] = bq;
        for (int kf = 0; kf < 4; ++kf) {
            v16h a  = load_afrag_lds(MB + kf * 32, OSTR, lane);
            v16h bw = load_bfrag_g(FCW, n0, 128, kf * 32, lane);
            acc     = wmma_f16(a, bw, acc);
        }
#pragma unroll
        for (int v = 0; v < 8; ++v) {
            int m = v + 8 * hi;
            out[(b0 + m) * 128 + n] = acc[v];
        }
    }
}

// ---------------------------------------------------------------------------
// Host launcher
// ---------------------------------------------------------------------------
extern "C" void kernel_launch(void* const* d_in, const int* in_sizes, int n_in,
                              void* d_out, int out_size, void* d_ws,
                              size_t ws_size, hipStream_t stream) {
    const float* M = (const float*)d_in[0];
    const int Btot = in_sizes[0] / 70;

    // workspace carve-up
    char* w = (char*)d_ws;
    auto alloc = [&](size_t bytes) {
        void* p = (void*)w;
        w += (bytes + 255) & ~(size_t)255;
        return p;
    };
    float*     X    = (float*)alloc((size_t)Btot * 210 * sizeof(float));
    _Float16*  W1ih = (_Float16*)alloc(2 * 256 * 32 * sizeof(_Float16));
    _Float16*  W1hh = (_Float16*)alloc(2 * 256 * 64 * sizeof(_Float16));
    float*     B1   = (float*)alloc(2 * 256 * sizeof(float));
    _Float16*  W2ih = (_Float16*)alloc(2 * 256 * 128 * sizeof(_Float16));
    _Float16*  W2hh = (_Float16*)alloc(2 * 256 * 64 * sizeof(_Float16));
    float*     B2   = (float*)alloc(2 * 256 * sizeof(float));
    _Float16*  A1Wh = (_Float16*)alloc(128 * 128 * sizeof(_Float16));
    _Float16*  A2Wh = (_Float16*)alloc(128 * 128 * sizeof(_Float16));
    _Float16*  FCWh = (_Float16*)alloc(128 * 128 * sizeof(_Float16));

    PrepArgs pa;
    pa.l1Wih[0] = (const float*)d_in[15]; pa.l1Whh[0] = (const float*)d_in[16];
    pa.l1bih[0] = (const float*)d_in[17]; pa.l1bhh[0] = (const float*)d_in[18];
    pa.l1Wih[1] = (const float*)d_in[19]; pa.l1Whh[1] = (const float*)d_in[20];
    pa.l1bih[1] = (const float*)d_in[21]; pa.l1bhh[1] = (const float*)d_in[22];
    pa.l2Wih[0] = (const float*)d_in[23]; pa.l2Whh[0] = (const float*)d_in[24];
    pa.l2bih[0] = (const float*)d_in[25]; pa.l2bhh[0] = (const float*)d_in[26];
    pa.l2Wih[1] = (const float*)d_in[27]; pa.l2Whh[1] = (const float*)d_in[28];
    pa.l2bih[1] = (const float*)d_in[29]; pa.l2bhh[1] = (const float*)d_in[30];
    pa.a1W = (const float*)d_in[31];
    pa.a2W = (const float*)d_in[33];
    pa.fcW = (const float*)d_in[35];
    pa.W1ih = W1ih; pa.W1hh = W1hh; pa.B1 = B1;
    pa.W2ih = W2ih; pa.W2hh = W2hh; pa.B2 = B2;
    pa.A1W = A1Wh;  pa.A2W = A2Wh;  pa.FCW = FCWh;

    prep_kernel<<<64, 256, 0, stream>>>(pa);

    graph_kernel<<<Btot / 8, 256, 0, stream>>>(
        M, (const float*)d_in[1], (const float*)d_in[2], (const float*)d_in[3],
        (const float*)d_in[4], (const float*)d_in[5], (const float*)d_in[6],
        (const float*)d_in[7], (const float*)d_in[8], (const float*)d_in[9],
        (const float*)d_in[10], (const float*)d_in[11], (const float*)d_in[12],
        (const float*)d_in[13], (const float*)d_in[14], X);

    fused_seq_kernel<<<Btot / 16, 256, 0, stream>>>(
        X, W1ih, W1hh, B1, W2ih, W2hh, B2, A1Wh, (const float*)d_in[32], A2Wh,
        (const float*)d_in[34], FCWh, (const float*)d_in[36], (float*)d_out);
}
